// RNN_28260884808397
// MI455X (gfx1250) — matrix operand order repaired and checked by reading, not compiled, not run on hardware
//
#include <hip/hip_runtime.h>
#include <hip/hip_bf16.h>

#define TLEN 4096

typedef __attribute__((ext_vector_type(16))) _Float16 v16h;
typedef __attribute__((ext_vector_type(8)))  _Float16 v8h;
typedef __attribute__((ext_vector_type(8)))  float    v8f;

__device__ __forceinline__ v16h combine16(v8h lo, v8h hi) {
    v16h r;
#pragma unroll
    for (int i = 0; i < 8; ++i) { r[i] = lo[i]; r[i + 8] = hi[i]; }
    return r;
}

// A-fragment (16x32 f16) for V_WMMA_F32_16X16X32_F16, from row-major f16 [16][ldk].
// Lane l holds row m=l%16; lanes 0-15: K = kb*32 + {0..15}; lanes 16-31: +16.
__device__ __forceinline__ v16h load_a_frag(const _Float16* base, int lane, int kb, int ldk) {
    const int m    = lane & 15;
    const int koff = kb * 32 + ((lane & 16) ? 16 : 0);
    const _Float16* p = base + m * ldk + koff;
    v8h lo = *(const v8h*)(p);        // K = koff .. koff+7
    v8h hi = *(const v8h*)(p + 8);    // K = koff+8 .. koff+15
    return combine16(lo, hi);
}

// B-fragment (32x16 f16) where B[k][n] = W[n][k], W row-major f16 [n][ldk].
// Lane l holds column n=n0+l%16; contiguous K-run of 16 halfs = contiguous row chunk of W.
__device__ __forceinline__ v16h load_b_frag_lds(const _Float16* W, int lane, int n0, int kb, int ldk) {
    const int n     = n0 + (lane & 15);
    const int kbase = kb * 32 + ((lane & 16) ? 16 : 0);
    const _Float16* p = W + n * ldk + kbase;
    v8h lo = *(const v8h*)(p);
    v8h hi = *(const v8h*)(p + 8);
    return combine16(lo, hi);
}

// Same B-fragment but sourced from f32 global weights row-major [n][ldk] (converted to f16).
__device__ __forceinline__ v16h load_b_frag_f32(const float* W, int lane, int n0, int kb, int ldk) {
    const int n     = n0 + (lane & 15);
    const int kbase = kb * 32 + ((lane & 16) ? 16 : 0);
    const float* p = W + n * ldk + kbase;
    v16h r;
#pragma unroll
    for (int j = 0; j < 16; ++j) r[j] = (_Float16)p[j];
    return r;
}

// ---------------------------------------------------------------------------
// Kernel 1: x_proj = input @ W1x.T + b1   (fully parallel; f16 WMMA, f32 acc)
// Each workgroup: 128 rows of flattened (B*T, 128) input; 8 waves, wave w owns
// M-tile w (16 rows) and loops the 8 N-tiles.
// ---------------------------------------------------------------------------
__global__ void __launch_bounds__(256) xproj_kernel(const float* __restrict__ in,
                                                    const float* __restrict__ W1,
                                                    const float* __restrict__ b1,
                                                    float* __restrict__ xp) {
    __shared__ __align__(16) _Float16 As[128 * 128];   // input block, f16
    __shared__ __align__(16) _Float16 Ws[128 * 128];   // W1x rows,   f16
    const int lane = threadIdx.x & 31;
    const int wave = threadIdx.x >> 5;
    const size_t row0 = (size_t)blockIdx.x * 128;

    // Stage 128x128 input block f32 -> f16 (coalesced float4 loads).
    const float* src = in + row0 * 128;
#pragma unroll
    for (int it = 0; it < 16; ++it) {
        const int idx4 = it * 256 + threadIdx.x;           // 4096 float4 total
        const float4 v = ((const float4*)src)[idx4];
        _Float16* d = &As[idx4 * 4];
        d[0] = (_Float16)v.x; d[1] = (_Float16)v.y;
        d[2] = (_Float16)v.z; d[3] = (_Float16)v.w;
    }
    // Stage W1x = W1[:, 128:256] rows as f16 (row-major, stride 256 in W1).
#pragma unroll
    for (int it = 0; it < 16; ++it) {
        const int idx4 = it * 256 + threadIdx.x;
        const int nrow = idx4 >> 5;                        // 32 float4 per row
        const int c    = (idx4 & 31) * 4;
        const float4 v = *(const float4*)(W1 + nrow * 256 + 128 + c);
        _Float16* d = &Ws[nrow * 128 + c];
        d[0] = (_Float16)v.x; d[1] = (_Float16)v.y;
        d[2] = (_Float16)v.z; d[3] = (_Float16)v.w;
    }
    __syncthreads();

    v16h af[4];
#pragma unroll
    for (int kb = 0; kb < 4; ++kb)
        af[kb] = load_a_frag(As + wave * 16 * 128, lane, kb, 128);

    const int mlo = (lane & 16) ? 8 : 0;
#pragma unroll
    for (int nt = 0; nt < 8; ++nt) {
        const float b1v = b1[nt * 16 + (lane & 15)];
        v8f acc = {};
#pragma unroll
        for (int kb = 0; kb < 4; ++kb) {
            v16h bf = load_b_frag_lds(Ws, lane, nt * 16, kb, 128);
            acc = __builtin_amdgcn_wmma_f32_16x16x32_f16(false, af[kb], false, bf,
                                                         (short)0, acc, false, false);
        }
        float* o = xp + (row0 + wave * 16 + mlo) * 128 + nt * 16 + (lane & 15);
#pragma unroll
        for (int i = 0; i < 8; ++i) o[(size_t)i * 128] = acc[i] + b1v;
    }
}

// ---------------------------------------------------------------------------
// Kernel 2: sequential scan over T. 4 workgroups x 16 batch rows; 8 waves,
// wave w owns column tile w.
//  - W1h/W2 B-fragments pinned in registers for all 4096 steps.
//  - ONE barrier per step (ping-pong h buffer).
//  - A-fragments of h_t loaded once, feeding both GEMM2(t) and GEMM1(t+1);
//    GEMM2 + output stores are off the serial critical path.
//  - GEMM1 split into two accumulator chains to halve dependent-WMMA latency.
//  - xp loads software-pipelined one full step ahead (L2-resident workspace).
// ---------------------------------------------------------------------------
__global__ void __launch_bounds__(256) rnn_scan_kernel(const float* __restrict__ xp,
                                                       const float* __restrict__ W1,
                                                       const float* __restrict__ W2,
                                                       const float* __restrict__ b2,
                                                       float* __restrict__ out) {
    __shared__ __align__(16) _Float16 hbuf[2][16 * 128];
    const int lane = threadIdx.x & 31;
    const int wave = threadIdx.x >> 5;
    const int n0   = wave * 16;
    const int mlo  = (lane & 16) ? 8 : 0;
    const int nn   = n0 + (lane & 15);

    // Persistent weight fragments: W1h = W1[:, :128] (row stride 256), W2 (stride 128).
    v16h w1f[4], w2f[4];
#pragma unroll
    for (int kb = 0; kb < 4; ++kb) {
        w1f[kb] = load_b_frag_f32(W1, lane, n0, kb, 256);
        w2f[kb] = load_b_frag_f32(W2, lane, n0, kb, 128);
    }
    const float b2v = b2[nn];

    // Per-lane streaming pointers into xp / out: row r, fixed column nn,
    // advancing by 128 floats per timestep.
    const float* xptr[8];
    float*       optr[8];
#pragma unroll
    for (int i = 0; i < 8; ++i) {
        const size_t r = (size_t)(blockIdx.x * 16 + mlo + i);
        xptr[i] = xp  + r * (size_t)TLEN * 128 + nn;
        optr[i] = out + r * (size_t)TLEN * 128 + nn;
    }

    // h_0 = 0 lives in registers: A-fragments start as zeros (no LDS init needed).
    v16h a[4] = {};

    // Pre-load xp for t=0 (pipeline prologue).
    float xv[8];
#pragma unroll
    for (int i = 0; i < 8; ++i) xv[i] = xptr[i][0];

    int p = 0;
    for (int t = 0; t < TLEN; ++t) {
        // Issue next step's xp loads a full iteration early.
        const int tn = (t + 1 < TLEN) ? (t + 1) : t;
        float xn[8];
#pragma unroll
        for (int i = 0; i < 8; ++i) xn[i] = xptr[i][(size_t)tn * 128];
        __builtin_prefetch(xptr[0] + (size_t)(t + 8) * 128, 0, 0);

        // GEMM1: h_{t-1} @ W1h.T  (two independent accumulator chains)
        v8f acc_a = {}, acc_b = {};
        acc_a = __builtin_amdgcn_wmma_f32_16x16x32_f16(false, a[0], false, w1f[0],
                                                       (short)0, acc_a, false, false);
        acc_b = __builtin_amdgcn_wmma_f32_16x16x32_f16(false, a[1], false, w1f[1],
                                                       (short)0, acc_b, false, false);
        acc_a = __builtin_amdgcn_wmma_f32_16x16x32_f16(false, a[2], false, w1f[2],
                                                       (short)0, acc_a, false, false);
        acc_b = __builtin_amdgcn_wmma_f32_16x16x32_f16(false, a[3], false, w1f[3],
                                                       (short)0, acc_b, false, false);

        // h_t = relu(acc + xp_t), stored f16 into ping-pong buffer.
        _Float16* hb = &hbuf[p][0];
#pragma unroll
        for (int i = 0; i < 8; ++i) {
            float v = acc_a[i] + acc_b[i] + xv[i];
            v = v > 0.0f ? v : 0.0f;
            hb[(mlo + i) * 128 + nn] = (_Float16)v;
        }
        __syncthreads();                   // h_t fully written & visible

        // One A-fragment load of h_t serves GEMM2(t) AND GEMM1(t+1).
#pragma unroll
        for (int kb = 0; kb < 4; ++kb) a[kb] = load_a_frag(hb, lane, kb, 128);

        // GEMM2: out_t = relu(h_t @ W2.T + b2)   (off the serial path)
        v8f acc2 = {};
#pragma unroll
        for (int kb = 0; kb < 4; ++kb)
            acc2 = __builtin_amdgcn_wmma_f32_16x16x32_f16(false, a[kb], false, w2f[kb],
                                                          (short)0, acc2, false, false);
#pragma unroll
        for (int i = 0; i < 8; ++i) {
            const float v = acc2[i] + b2v;
            optr[i][(size_t)t * 128] = v > 0.0f ? v : 0.0f;
        }

#pragma unroll
        for (int i = 0; i < 8; ++i) xv[i] = xn[i];
        p ^= 1;
    }
}

extern "C" void kernel_launch(void* const* d_in, const int* in_sizes, int n_in,
                              void* d_out, int out_size, void* d_ws, size_t ws_size,
                              hipStream_t stream) {
    const float* input = (const float*)d_in[0];   // (B, T, 128) f32
    const float* W1    = (const float*)d_in[1];   // (128, 256)  f32
    const float* b1    = (const float*)d_in[2];   // (128,)      f32
    const float* W2    = (const float*)d_in[3];   // (128, 128)  f32
    const float* b2    = (const float*)d_in[4];   // (128,)      f32
    float* out = (float*)d_out;                   // (B, T, 128) f32
    float* xp  = (float*)d_ws;                    // x_proj workspace: B*T*128 f32

    const int BT = in_sizes[0] / 128;             // B*T = 262144
    xproj_kernel<<<BT / 128, 256, 0, stream>>>(input, W1, b1, xp);

    const int BG = (BT / TLEN) / 16;              // batch groups of 16 rows -> 4
    rnn_scan_kernel<<<BG, 256, 0, stream>>>(xp, W1, W2, b2, out);
}
